// KITRO_86715389706286
// MI455X (gfx1250) — matrix operand order repaired and checked by reading, not compiled, not run on hardware
//
#include <hip/hip_runtime.h>
#include <hip/hip_bf16.h>

// ---------------------------------------------------------------------------
// Problem sizes (fixed by the reference)
// ---------------------------------------------------------------------------
#define BATCH   8192
#define NJ      25
#define FD      512
#define HD      1024
#define H2      512
#define NB      16
#define NROWS   (BATCH * NJ)          // 204800, divisible by 32
#define RPB     32                    // rows per block (2 M-tiles)
#define EPSV    1e-8f

typedef __bf16 bf16;
typedef __attribute__((ext_vector_type(16))) __bf16 v16bf;
typedef __attribute__((ext_vector_type(8)))  __bf16 v8bf;
typedef __attribute__((ext_vector_type(8)))  float  v8f;

__constant__ int PAR[NB] = {0, 0, 0, 0, 0, 0, 5, 7, 6, 8, 5, 6, 11, 13, 12, 14};
__constant__ int CHI[NB] = {1, 2, 3, 4, 5, 6, 7, 9, 8, 10, 11, 12, 13, 15, 14, 16};

static __device__ __forceinline__ v8f wmma_bf16(v16bf a, v16bf b, v8f c) {
  return __builtin_amdgcn_wmma_f32_16x16x32_bf16(false, a, false, b, (short)0, c,
                                                 false, false);
}

// ---------------------------------------------------------------------------
// CDNA5 async global->LDS DMA path (ASYNCcnt), with sync fallback.
// Builtin prototype (from clang diagnostic):
//   void __builtin_amdgcn_global_load_async_to_lds_b128(
//       int __vector(4) AS1* src, int __vector(4) AS3* dst,
//       imm int offset, imm int cpol)
// ---------------------------------------------------------------------------
#if defined(__gfx1250__) &&                                                \
    __has_builtin(__builtin_amdgcn_global_load_async_to_lds_b128) &&       \
    __has_builtin(__builtin_amdgcn_s_wait_asynccnt)
#define HAS_ASYNC 1
typedef int v4i_ __attribute__((vector_size(16)));
typedef __attribute__((address_space(1))) v4i_* gv4p;
typedef __attribute__((address_space(3))) v4i_* lv4p;
static __device__ __forceinline__ void async_copy16(const void* g, void* l) {
  __builtin_amdgcn_global_load_async_to_lds_b128((gv4p)g, (lv4p)l, 0, 0);
}
#else
#define HAS_ASYNC 0
#endif

// ---------------------------------------------------------------------------
// Kernel 0: one-shot weight conversion f32 -> bf16 into workspace.
//   W1bf: [512][1024], W2bf: [1024][512], C1bf: cW1 rows 4..515 -> [512][32],
//   C2bf: [32][64]
// ---------------------------------------------------------------------------
#define NW1 (FD * HD)            // 524288
#define NW2 (HD * H2)            // 524288
#define NC1 (FD * 32)            // 16384
#define NC2 (32 * 64)            // 2048
#define NPREP (NW1 + NW2 + NC1 + NC2)  // 1067008 = 4168 * 256

__global__ void prep_weights(const float* __restrict__ dW1,
                             const float* __restrict__ dW2,
                             const float* __restrict__ cW1,
                             const float* __restrict__ cW2,
                             bf16* __restrict__ W1bf, bf16* __restrict__ W2bf,
                             bf16* __restrict__ C1bf, bf16* __restrict__ C2bf) {
  int i = blockIdx.x * blockDim.x + threadIdx.x;
  if (i < NW1) {
    W1bf[i] = (bf16)dW1[i];
  } else if (i < NW1 + NW2) {
    int r = i - NW1;
    W2bf[r] = (bf16)dW2[r];
  } else if (i < NW1 + NW2 + NC1) {
    int r = i - NW1 - NW2;
    C1bf[r] = (bf16)cW1[(4 + (r >> 5)) * 32 + (r & 31)];  // skip dir/len rows
  } else if (i < NPREP) {
    int r = i - NW1 - NW2 - NC1;
    C2bf[r] = (bf16)cW2[r];
  }
}

// ---------------------------------------------------------------------------
// Kernel 1: depth MLP.  One block = 32 rows (2 M-tiles), 8 waves.
//   Each wave: layer1 8 n-tiles x 2 M-tiles (B fragment reused 2x),
//              layer2 4 n-tiles x 2 M-tiles.
// ---------------------------------------------------------------------------
__launch_bounds__(256) __global__
void depth_mlp(const float* __restrict__ feat,
               const float* __restrict__ db1, const float* __restrict__ db2,
               const float* __restrict__ dW3, const float* __restrict__ db3,
               const bf16* __restrict__ W1bf, const bf16* __restrict__ W2bf,
               float* __restrict__ depths) {
  // LDS layout (97KB):
  //   [0,   64K): H1 (32x1024 bf16); first reused as f32 staging (32x512 f32)
  //   [64K, 96K): Xbf (32x512 bf16); later reused as H2 (32x512 bf16)
  //   [96K, 97K): reduction scratch
  __shared__ __align__(16) char smem[64 * 1024 + 32 * 1024 + 256 * 4];
  bf16*  H1  = (bf16*)smem;
  float* Xf  = (float*)smem;                  // alias of H1 region (staging)
  bf16*  Xbf = (bf16*)(smem + 64 * 1024);
  bf16*  Hs2 = Xbf;                           // alias (after layer-1 reads)
  float* red = (float*)(smem + 96 * 1024);

  const int tid  = threadIdx.x;
  const int lane = tid & 31;
  const int wid  = tid >> 5;
  const long row0 = (long)blockIdx.x * RPB;
  const float* src = feat + row0 * FD;

  // ---- stage features -> LDS bf16 ----
#if HAS_ASYNC
  // DMA raw f32 tile into LDS (ASYNCcnt), then convert LDS f32 -> LDS bf16.
  for (int c = tid; c < RPB * FD / 4; c += 256)   // 4096 x 16B chunks
    async_copy16(src + c * 4, Xf + c * 4);
  __builtin_amdgcn_s_wait_asynccnt(0);
  __syncthreads();
  for (int i = tid; i < RPB * FD / 4; i += 256) {
    const float4 v = ((const float4*)Xf)[i];
    bf16* d = &Xbf[i * 4];
    d[0] = (bf16)v.x; d[1] = (bf16)v.y; d[2] = (bf16)v.z; d[3] = (bf16)v.w;
  }
#else
  for (int i = tid; i < RPB * FD / 4; i += 256) {
    const float4 v = ((const float4*)src)[i];
    bf16* d = &Xbf[i * 4];
    d[0] = (bf16)v.x; d[1] = (bf16)v.y; d[2] = (bf16)v.z; d[3] = (bf16)v.w;
  }
#endif
  __syncthreads();

  const int am  = lane & 15;            // A-fragment row (M within tile)
  const int sel = (lane >> 4) * 8;      // A-fragment K sub-select
  const int cn  = lane & 15;            // C-layout N within tile
  const int cmh = (lane >> 4) * 8;      // C-layout M base

  // ---- layer 1: N=1024 -> 64 tiles, 8 per wave x 2 M-tiles; K=512 ----
  v8f acc1[2][8];
#pragma unroll
  for (int j = 0; j < 8; ++j) {
    const float bv = db1[(wid * 8 + j) * 16 + cn];
#pragma unroll
    for (int v = 0; v < 8; ++v) { acc1[0][j][v] = bv; acc1[1][j][v] = bv; }
  }
  for (int k = 0; k < 16; ++k) {
    v16bf a0, a1;
    {
      const v8bf lo0 = *(const v8bf*)&Xbf[am * FD + k * 32 + sel];
      const v8bf hi0 = *(const v8bf*)&Xbf[am * FD + k * 32 + 16 + sel];
      const v8bf lo1 = *(const v8bf*)&Xbf[(16 + am) * FD + k * 32 + sel];
      const v8bf hi1 = *(const v8bf*)&Xbf[(16 + am) * FD + k * 32 + 16 + sel];
#pragma unroll
      for (int i = 0; i < 8; ++i) {
        a0[i] = lo0[i]; a0[i + 8] = hi0[i];
        a1[i] = lo1[i]; a1[i + 8] = hi1[i];
      }
    }
    const bf16* wrow = W1bf + (size_t)(k * 32 + sel) * HD + cn;  // lane = N col, elems = K (table order)
#pragma unroll
    for (int j = 0; j < 8; ++j) {
      v16bf bfr;
#pragma unroll
      for (int i = 0; i < 8; ++i) {
        bfr[i] = wrow[(size_t)i * HD + (wid * 8 + j) * 16];
        bfr[i + 8] = wrow[(size_t)(16 + i) * HD + (wid * 8 + j) * 16];
      }
      acc1[0][j] = wmma_bf16(a0, bfr, acc1[0][j]);   // B fragment reused 2x
      acc1[1][j] = wmma_bf16(a1, bfr, acc1[1][j]);
    }
  }
#pragma unroll
  for (int j = 0; j < 8; ++j) {
    const int nn = (wid * 8 + j) * 16 + cn;
#pragma unroll
    for (int t = 0; t < 2; ++t)
#pragma unroll
      for (int v = 0; v < 8; ++v) {
        float g = acc1[t][j][v];
        g = g > 0.f ? g : 0.f;
        H1[(t * 16 + cmh + v) * HD + nn] = (bf16)g;
      }
  }
  __syncthreads();

  // ---- layer 2: N=512 -> 32 tiles, 4 per wave x 2 M-tiles; K=1024 ----
  v8f acc2[2][4];
#pragma unroll
  for (int j = 0; j < 4; ++j) {
    const float bv = db2[(wid * 4 + j) * 16 + cn];
#pragma unroll
    for (int v = 0; v < 8; ++v) { acc2[0][j][v] = bv; acc2[1][j][v] = bv; }
  }
  for (int k = 0; k < 32; ++k) {
    v16bf a0, a1;
    {
      const v8bf lo0 = *(const v8bf*)&H1[am * HD + k * 32 + sel];
      const v8bf hi0 = *(const v8bf*)&H1[am * HD + k * 32 + 16 + sel];
      const v8bf lo1 = *(const v8bf*)&H1[(16 + am) * HD + k * 32 + sel];
      const v8bf hi1 = *(const v8bf*)&H1[(16 + am) * HD + k * 32 + 16 + sel];
#pragma unroll
      for (int i = 0; i < 8; ++i) {
        a0[i] = lo0[i]; a0[i + 8] = hi0[i];
        a1[i] = lo1[i]; a1[i + 8] = hi1[i];
      }
    }
    const bf16* wrow = W2bf + (size_t)(k * 32 + sel) * H2 + cn;
#pragma unroll
    for (int j = 0; j < 4; ++j) {
      v16bf bfr;
#pragma unroll
      for (int i = 0; i < 8; ++i) {
        bfr[i] = wrow[(size_t)i * H2 + (wid * 4 + j) * 16];
        bfr[i + 8] = wrow[(size_t)(16 + i) * H2 + (wid * 4 + j) * 16];
      }
      acc2[0][j] = wmma_bf16(a0, bfr, acc2[0][j]);
      acc2[1][j] = wmma_bf16(a1, bfr, acc2[1][j]);
    }
  }
#pragma unroll
  for (int j = 0; j < 4; ++j) {
    const int nn = (wid * 4 + j) * 16 + cn;
#pragma unroll
    for (int t = 0; t < 2; ++t)
#pragma unroll
      for (int v = 0; v < 8; ++v) {
        float g = acc2[t][j][v];
        g = g > 0.f ? g : 0.f;
        Hs2[(t * 16 + cmh + v) * H2 + nn] = (bf16)g;
      }
  }
  __syncthreads();

  // ---- layer 3: depth = Hs2 @ W3 + b3 (N=1, VALU dot + LDS reduce) ----
  {
    const int m = tid >> 3;     // 32 rows
    const int part = tid & 7;   // 8 partials per row
    float s = 0.f;
    for (int k = part * 64; k < part * 64 + 64; ++k)
      s += (float)Hs2[m * H2 + k] * dW3[k];
    red[tid] = s;
    __syncthreads();
    if (part == 0) {
      float t = 0.f;
#pragma unroll
      for (int i = 0; i < 8; ++i) t += red[m * 8 + i];
      depths[row0 + m] = t + db3[0];
    }
  }
}

// ---------------------------------------------------------------------------
// Kernel 2: bone refinement.  One wave per batch item, 8 waves/block.
//   avg_feat @ cW1[4:] is hoisted (WMMA, K=512, M = 16 bones); 3 iterations
//   update only the 4-wide [dir,len] correction + small WMMA layer.
// ---------------------------------------------------------------------------
__launch_bounds__(256) __global__
void refine(const float* __restrict__ poses2d, const float* __restrict__ feat,
            const float* __restrict__ conf,
            const float* __restrict__ cW1, const float* __restrict__ cb1,
            const float* __restrict__ cb2,
            const float* __restrict__ cW3, const float* __restrict__ cb3,
            const bf16* __restrict__ C1bf, const bf16* __restrict__ C2bf,
            const float* __restrict__ depths, float* __restrict__ out) {
  __shared__ float DL[8][NB][4];       // dir(3)+len(1) per bone, per wave
  __shared__ bf16  AB[8][NB * 32];     // g1 in A-friendly row-major layout
  __shared__ float G2[8][NB * 64];     // g2 activations

  const int lane = threadIdx.x & 31;
  const int wid  = threadIdx.x >> 5;
  const int b    = blockIdx.x * 8 + wid;

  const int am  = lane & 15;           // bone index for A fragments
  const int sel = (lane >> 4) * 8;
  const int cn  = lane & 15;
  const int cmh = (lane >> 4) * 8;

  // ---- invariant projection P = avg_feat @ cW1[4:,:] + cb1  (N=32) ----
  v8f P0, P1;
  {
    const float b0 = cb1[cn];
    const float b1 = cb1[16 + cn];
#pragma unroll
    for (int v = 0; v < 8; ++v) { P0[v] = b0; P1[v] = b1; }
  }
  const int pj = PAR[am];
  const int cj = CHI[am];
  {
    const float* fp = feat + ((size_t)b * NJ + pj) * FD;
    const float* fc = feat + ((size_t)b * NJ + cj) * FD;
    for (int k = 0; k < 16; ++k) {
      const int o1 = k * 32 + sel;
      const int o2 = o1 + 16;
      const float4 pa = *(const float4*)&fp[o1];
      const float4 pb = *(const float4*)&fp[o1 + 4];
      const float4 ca = *(const float4*)&fc[o1];
      const float4 cb = *(const float4*)&fc[o1 + 4];
      const float4 pc = *(const float4*)&fp[o2];
      const float4 pd = *(const float4*)&fp[o2 + 4];
      const float4 cc = *(const float4*)&fc[o2];
      const float4 cd = *(const float4*)&fc[o2 + 4];
      v16bf a;
      a[0]  = (bf16)(0.5f * (pa.x + ca.x)); a[1]  = (bf16)(0.5f * (pa.y + ca.y));
      a[2]  = (bf16)(0.5f * (pa.z + ca.z)); a[3]  = (bf16)(0.5f * (pa.w + ca.w));
      a[4]  = (bf16)(0.5f * (pb.x + cb.x)); a[5]  = (bf16)(0.5f * (pb.y + cb.y));
      a[6]  = (bf16)(0.5f * (pb.z + cb.z)); a[7]  = (bf16)(0.5f * (pb.w + cb.w));
      a[8]  = (bf16)(0.5f * (pc.x + cc.x)); a[9]  = (bf16)(0.5f * (pc.y + cc.y));
      a[10] = (bf16)(0.5f * (pc.z + cc.z)); a[11] = (bf16)(0.5f * (pc.w + cc.w));
      a[12] = (bf16)(0.5f * (pd.x + cd.x)); a[13] = (bf16)(0.5f * (pd.y + cd.y));
      a[14] = (bf16)(0.5f * (pd.z + cd.z)); a[15] = (bf16)(0.5f * (pd.w + cd.w));
      const bf16* r = C1bf + (size_t)(k * 32 + sel) * 32 + cn;  // lane = N col, elems = K (table order)
      v16bf bf0, bf1;
#pragma unroll
      for (int i = 0; i < 8; ++i) {
        bf0[i] = r[i * 32]; bf0[i + 8] = r[(16 + i) * 32];
        bf1[i] = r[i * 32 + 16]; bf1[i + 8] = r[(16 + i) * 32 + 16];
      }
      P0 = wmma_bf16(a, bf0, P0);
      P1 = wmma_bf16(a, bf1, P1);
    }
  }

  // ---- load poses into lanes (lane == joint) ----
  float px = 0.f, py = 0.f, pz = 0.f, cf = 1.f;
  if (lane < NJ) {
    px = poses2d[((size_t)b * NJ + lane) * 2];
    py = poses2d[((size_t)b * NJ + lane) * 2 + 1];
    pz = depths[(size_t)b * NJ + lane];
    cf = conf[(size_t)b * NJ + lane];
  }

  for (int it = 0; it < 3; ++it) {
    // bone dir/len via cross-lane gathers (parent/child joints < 17)
    const float pxp = __shfl(px, pj), pyp = __shfl(py, pj), pzp = __shfl(pz, pj);
    const float pxc = __shfl(px, cj), pyc = __shfl(py, cj), pzc = __shfl(pz, cj);
    const float vx = pxc - pxp, vy = pyc - pyp, vz = pzc - pzp;
    const float len = sqrtf(vx * vx + vy * vy + vz * vz);
    const float inv = 1.f / (len + EPSV);
    if (lane < NB) {
      DL[wid][am][0] = vx * inv;
      DL[wid][am][1] = vy * inv;
      DL[wid][am][2] = vz * inv;
      DL[wid][am][3] = len;
    }
    __syncthreads();

    // g1 = relu(P + [dir,len] @ cW1[0:4,:]) -> AB (bf16, row-major [m][32])
#pragma unroll
    for (int nt = 0; nt < 2; ++nt) {
      const int n = nt * 16 + cn;
      const float w0 = cW1[0 * 32 + n], w1 = cW1[1 * 32 + n];
      const float w2 = cW1[2 * 32 + n], w3 = cW1[3 * 32 + n];
#pragma unroll
      for (int v = 0; v < 8; ++v) {
        const int mm = cmh + v;
        const float4 dl = *(const float4*)&DL[wid][mm][0];
        float g = (nt ? P1[v] : P0[v]) + dl.x * w0 + dl.y * w1 + dl.z * w2 + dl.w * w3;
        g = g > 0.f ? g : 0.f;
        AB[wid][mm * 32 + n] = (bf16)g;
      }
    }
    __syncthreads();

    // g2 = relu(g1 @ cW2 + cb2)   (K=32, one wmma per 16-col tile, N=64)
    v16bf a;
    {
      const v8bf lo = *(const v8bf*)&AB[wid][am * 32 + sel];
      const v8bf hi = *(const v8bf*)&AB[wid][am * 32 + 16 + sel];
#pragma unroll
      for (int i = 0; i < 8; ++i) { a[i] = lo[i]; a[i + 8] = hi[i]; }
    }
#pragma unroll
    for (int nt = 0; nt < 4; ++nt) {
      v8f acc;
      const float bv = cb2[nt * 16 + cn];
#pragma unroll
      for (int v = 0; v < 8; ++v) acc[v] = bv;
      v16bf bfr;
#pragma unroll
      for (int i = 0; i < 8; ++i) {
        bfr[i] = C2bf[(sel + i) * 64 + nt * 16 + cn];
        bfr[i + 8] = C2bf[(16 + sel + i) * 64 + nt * 16 + cn];
      }
      acc = wmma_bf16(a, bfr, acc);
#pragma unroll
      for (int v = 0; v < 8; ++v) {
        float g = acc[v];
        g = g > 0.f ? g : 0.f;
        G2[wid][(cmh + v) * 64 + nt * 16 + cn] = g;
      }
    }
    __syncthreads();

    // pose_update = mean_m( g2 @ cW3 + cb3 )  via column sums + wave reduce
    float u0 = 0.f, u1 = 0.f, u2 = 0.f;
    for (int kk = lane; kk < 64; kk += 32) {
      float cs = 0.f;
#pragma unroll
      for (int mm = 0; mm < NB; ++mm) cs += G2[wid][mm * 64 + kk];
      u0 += cs * cW3[kk * 3 + 0];
      u1 += cs * cW3[kk * 3 + 1];
      u2 += cs * cW3[kk * 3 + 2];
    }
#pragma unroll
    for (int off = 16; off >= 1; off >>= 1) {
      u0 += __shfl_xor(u0, off);
      u1 += __shfl_xor(u1, off);
      u2 += __shfl_xor(u2, off);
    }
    u0 = u0 * (1.f / NB) + cb3[0];
    u1 = u1 * (1.f / NB) + cb3[1];
    u2 = u2 * (1.f / NB) + cb3[2];

    px = (px + 0.1f * u0) * cf;
    py = (py + 0.1f * u1) * cf;
    pz = (pz + 0.1f * u2) * cf;
    __syncthreads();  // protect DL/AB/G2 for next iteration
  }

  if (lane < NJ) {
    const size_t o = ((size_t)b * NJ + lane) * 3;
    out[o] = px; out[o + 1] = py; out[o + 2] = pz;
  }
}

// ---------------------------------------------------------------------------
// Launch
// ---------------------------------------------------------------------------
extern "C" void kernel_launch(void* const* d_in, const int* in_sizes, int n_in,
                              void* d_out, int out_size, void* d_ws, size_t ws_size,
                              hipStream_t stream) {
  const float* poses2d = (const float*)d_in[0];
  const float* feat    = (const float*)d_in[1];
  const float* confid  = (const float*)d_in[2];
  const float* dW1 = (const float*)d_in[3];
  const float* db1 = (const float*)d_in[4];
  const float* dW2 = (const float*)d_in[5];
  const float* db2 = (const float*)d_in[6];
  const float* dW3 = (const float*)d_in[7];
  const float* db3 = (const float*)d_in[8];
  const float* cW1 = (const float*)d_in[9];
  const float* cb1 = (const float*)d_in[10];
  const float* cW2 = (const float*)d_in[11];
  const float* cb2 = (const float*)d_in[12];
  const float* cW3 = (const float*)d_in[13];
  const float* cb3 = (const float*)d_in[14];
  float* out = (float*)d_out;

  // workspace layout (bf16 weights + depths scratch)
  bf16* W1bf = (bf16*)d_ws;            // 524288
  bf16* W2bf = W1bf + NW1;             // 524288
  bf16* C1bf = W2bf + NW2;             // 16384
  bf16* C2bf = C1bf + NC1;             // 2048
  float* depths = (float*)(C2bf + NC2);  // 204800 floats (offset is 256B-aligned)

  prep_weights<<<NPREP / 256, 256, 0, stream>>>(dW1, dW2, cW1, cW2,
                                                W1bf, W2bf, C1bf, C2bf);
  depth_mlp<<<NROWS / RPB, 256, 0, stream>>>(feat, db1, db2, dW3, db3,
                                             W1bf, W2bf, depths);
  refine<<<BATCH / 8, 256, 0, stream>>>(poses2d, feat, confid,
                                        cW1, cb1, cb2, cW3, cb3,
                                        C1bf, C2bf, depths, out);
}
